// rnn_15290083573858
// MI455X (gfx1250) — matrix operand order repaired and checked by reading, not compiled, not run on hardware
//
#include <hip/hip_runtime.h>

#define T_STEPS 2048
#define BATCH   256
#define DIN     64
#define DH      128
#define XSTRIDE 72    // 64 + 8 pad (bf16 elems) -> conflict-optimal LDS b128 access
#define HSTRIDE 136   // 128 + 8 pad (bf16 elems)

typedef __attribute__((ext_vector_type(16))) __bf16 v16bf;
typedef __attribute__((ext_vector_type(8)))  __bf16 v8bf;
typedef __attribute__((ext_vector_type(8)))  float  v8f;

#if defined(__has_builtin)
#  if __has_builtin(__builtin_amdgcn_tanhf)
#    define HAS_HW_TANH 1
#  else
#    define HAS_HW_TANH 0
#  endif
#else
#  define HAS_HW_TANH 0
#endif

__device__ __forceinline__ float fast_tanh(float x) {
#if HAS_HW_TANH
    return __builtin_amdgcn_tanhf(x);   // single v_tanh_f32 on gfx1250
#else
    float cx = fminf(fmaxf(x, -15.0f), 15.0f);
    float e2 = __builtin_amdgcn_exp2f(cx * 2.8853900817779268f); // 2*log2(e)
    return (e2 - 1.0f) * __builtin_amdgcn_rcpf(e2 + 1.0f);
#endif
}

// A-fragment (16-bit, 16x32), per ISA 7.12.2: lane l holds row M=l&15;
// lanes 0-15 take K {kb..kb+7, kb+16..kb+23}, lanes 16-31 take
// {kb+8..kb+15, kb+24..kb+31}. Caller passes klo = kb + (lane>=16 ? 8 : 0).
// Source is row-major fp32 weight matrix W[N][K]; converted once to bf16.
__device__ __forceinline__ v16bf load_a_frag_w(const float* __restrict__ W, int K, int row, int klo) {
    v16bf r;
    const float* p0 = W + row * K + klo;
    const float* p1 = p0 + 16;
#pragma unroll
    for (int i = 0; i < 8; ++i) { r[i] = (__bf16)p0[i]; r[8 + i] = (__bf16)p1[i]; }
    return r;
}

// B-fragment (16-bit, 32x16), same K order as A: lane l (0-15) = column N=l
// holding K {kb..kb+7, kb+16..kb+23}; lanes 16-31 = column N=l-16 holding
// K {kb+8..kb+15, kb+24..kb+31}. Column N=m of x^T / h^T is row m of x / h
// -> two 16-byte runs in LDS. Caller passes klo = kb + (lane>=16 ? 8 : 0).
__device__ __forceinline__ v16bf lds_frag_b(const __bf16* s, int stride, int m, int klo) {
    union { v16bf v; v8bf h[2]; } u;
    u.h[0] = *(const v8bf*)(s + m * stride + klo);
    u.h[1] = *(const v8bf*)(s + m * stride + klo + 16);
    return u.v;
}

__global__ __launch_bounds__(256, 1)
void rnn_wmma_kernel(const float* __restrict__ x,
                     const float* __restrict__ W_ih,
                     const float* __restrict__ b_ih,
                     const float* __restrict__ W_hh,
                     const float* __restrict__ b_hh,
                     float* __restrict__ out) {
    // double-buffered x tile and hidden state (bf16, row-major, in LDS)
    __shared__ __align__(16) __bf16 s_x[2][16 * XSTRIDE];
    __shared__ __align__(16) __bf16 s_h[2][16 * HSTRIDE];

    const int tid   = threadIdx.x;
    const int lane  = tid & 31;
    const int wave  = tid >> 5;        // 0..7: owns hidden cols 16*wave..16*wave+15
    const int lhalf = lane >> 4;       // 0/1
    const int lmod  = lane & 15;
    const int akoff = lhalf * 8;       // A-fragment K offset per lane half
    const int bkoff = lhalf * 8;       // B-fragment K offset per lane half
    const int b0    = blockIdx.x * 16; // batch-row tile base

    // h_{-1} = 0 lives in s_h[1] (read at t=0); zero both for safety
    for (int i = tid; i < 16 * HSTRIDE; i += 256) {
        s_h[0][i] = (__bf16)0.0f;
        s_h[1][i] = (__bf16)0.0f;
    }

    // Transposed GEMM: D^T = W_tile x act^T  ->  weights are the A operand,
    // loop-invariant, held in VGPRs for the entire time loop.
    const int nrow = wave * 16 + lmod; // weight row this lane holds (A layout)
    v16bf aIh[2], aHh[4];
#pragma unroll
    for (int kb = 0; kb < 2; ++kb) aIh[kb] = load_a_frag_w(W_ih, DIN, nrow, kb * 32 + akoff);
#pragma unroll
    for (int kb = 0; kb < 4; ++kb) aHh[kb] = load_a_frag_w(W_hh, DH, nrow, kb * 32 + akoff);

    // D^T: lane = batch col m=lmod, VGPR r = hidden unit nb+r
    const int nb = wave * 16 + lhalf * 8;
    v8f biasvec;
#pragma unroll
    for (int r = 0; r < 8; ++r) biasvec[r] = b_ih[nb + r] + b_hh[nb + r];

    // x staging role: thread -> (row, 4 cols) of the 16x64 x_t tile
    const int xr = tid >> 4;
    const int xc = (tid & 15) * 4;
    const float* xbase = x + ((size_t)b0 + xr) * DIN + xc;

    // pre-stage x(0) into buffer 0
    {
        float4 xv = *(const float4*)xbase;
        __bf16* xd = &s_x[0][xr * XSTRIDE + xc];
        xd[0] = (__bf16)xv.x; xd[1] = (__bf16)xv.y;
        xd[2] = (__bf16)xv.z; xd[3] = (__bf16)xv.w;
    }

    float hout[8];

    // One time step. `buf` is always a literal at the call site so both LDS
    // buffers resolve to base+immediate ds offsets (no per-step address math).
    auto step = [&](int t, int buf) __attribute__((always_inline)) {
        const int nbuf = buf ^ 1;

        __syncthreads(); // x(t) in s_x[buf] and h(t-1) in s_h[nbuf] now visible

        // B fragments (activations, transposed): two 16B runs per lane
        v16bf bX0 = lds_frag_b(s_x[buf], XSTRIDE, lmod,  0 + bkoff);
        v16bf bX1 = lds_frag_b(s_x[buf], XSTRIDE, lmod, 32 + bkoff);
        v16bf bH0 = lds_frag_b(s_h[nbuf], HSTRIDE, lmod,  0 + bkoff);
        v16bf bH1 = lds_frag_b(s_h[nbuf], HSTRIDE, lmod, 32 + bkoff);
        v16bf bH2 = lds_frag_b(s_h[nbuf], HSTRIDE, lmod, 64 + bkoff);
        v16bf bH3 = lds_frag_b(s_h[nbuf], HSTRIDE, lmod, 96 + bkoff);

        // overlap next step's x fetch with this step's math
        if (t + 1 < T_STEPS) {
            const float* xpt = xbase + ((size_t)(t + 1)) * BATCH * DIN;
            float4 xv = *(const float4*)xpt;
            __builtin_prefetch(xpt + BATCH * DIN, 0, 3); // t+2, global_prefetch_b8
            __bf16* xd = &s_x[nbuf][xr * XSTRIDE + xc];
            xd[0] = (__bf16)xv.x; xd[1] = (__bf16)xv.y;
            xd[2] = (__bf16)xv.z; xd[3] = (__bf16)xv.w;
        }

        // Two INDEPENDENT accumulation chains (depth 3 each) so the hardware
        // can interleave dependent-WMMA latency; acc1 C starts as inline 0.
        v8f acc0 = biasvec, acc1 = {};
        acc0 = __builtin_amdgcn_wmma_f32_16x16x32_bf16(false, aIh[0], false, bX0, (short)0, acc0, false, false);
        acc1 = __builtin_amdgcn_wmma_f32_16x16x32_bf16(false, aIh[1], false, bX1, (short)0, acc1, false, false);
        acc0 = __builtin_amdgcn_wmma_f32_16x16x32_bf16(false, aHh[0], false, bH0, (short)0, acc0, false, false);
        acc1 = __builtin_amdgcn_wmma_f32_16x16x32_bf16(false, aHh[1], false, bH1, (short)0, acc1, false, false);
        acc0 = __builtin_amdgcn_wmma_f32_16x16x32_bf16(false, aHh[2], false, bH2, (short)0, acc0, false, false);
        acc1 = __builtin_amdgcn_wmma_f32_16x16x32_bf16(false, aHh[3], false, bH3, (short)0, acc1, false, false);

        // combine chains, tanh; lane's 8 results are h[m][nb..nb+7] ->
        // 8 consecutive bf16 = ONE ds_store_b128
        v8bf pk;
#pragma unroll
        for (int r = 0; r < 8; ++r) {
            float hv = fast_tanh(acc0[r] + acc1[r]);
            hout[r] = hv;
            pk[r] = (__bf16)hv;
        }
        *(v8bf*)(&s_h[buf][lmod * HSTRIDE + nb]) = pk;
    };

#pragma unroll 1
    for (int t = 0; t < T_STEPS; t += 2) {
        step(t,     0);   // literal buf -> immediate LDS offsets
        step(t + 1, 1);
    }

    // final hidden state in fp32: 8 consecutive floats per lane -> 2x b128
    float* op = out + ((size_t)(b0 + lmod)) * DH + nb;
    *(float4*)(op)     = make_float4(hout[0], hout[1], hout[2], hout[3]);
    *(float4*)(op + 4) = make_float4(hout[4], hout[5], hout[6], hout[7]);
}

extern "C" void kernel_launch(void* const* d_in, const int* in_sizes, int n_in,
                              void* d_out, int out_size, void* d_ws, size_t ws_size,
                              hipStream_t stream) {
    const float* x    = (const float*)d_in[0];
    const float* W_ih = (const float*)d_in[1];
    const float* b_ih = (const float*)d_in[2];
    const float* W_hh = (const float*)d_in[3];
    const float* b_hh = (const float*)d_in[4];
    rnn_wmma_kernel<<<dim3(BATCH / 16), dim3(256), 0, stream>>>(
        x, W_ih, b_ih, W_hh, b_hh, (float*)d_out);
}
